// GaussianImage_RS_29953101922994
// MI455X (gfx1250) — matrix operand order repaired and checked by reading, not compiled, not run on hardware
//
#include <hip/hip_runtime.h>
#include <math.h>

#define IMG_H 256
#define IMG_W 256
#define NGAUSS 2048
#define CHUNK 1024
#define WAVES_PER_BLOCK 8
#define TPB 256
#define PIX_PER_WAVE 32   // two 16-pixel WMMA tiles per wave

#if __has_builtin(__builtin_amdgcn_exp2f)
#define EXP2F __builtin_amdgcn_exp2f
#else
#define EXP2F exp2f
#endif

typedef __attribute__((ext_vector_type(16))) _Float16 v16h;
typedef __attribute__((ext_vector_type(8)))  float    v8f;

__global__ __launch_bounds__(TPB)
void gsplat_wmma_kernel(const float* __restrict__ xyz,
                        const float* __restrict__ scaling,
                        const float* __restrict__ rotation,
                        const float* __restrict__ features,
                        const float* __restrict__ opacity,
                        float* __restrict__ out)
{
    // SoA projected params: cx, cy, ca'(=0.5*ca/ln2), cb'(=cb/ln2), cc'(=0.5*cc/ln2), op
    __shared__ __align__(32) float    s_g [6][CHUNK];   // 24 KB
    __shared__ __align__(32) _Float16 s_ft[4][CHUNK];   //  8 KB, row 3 zero-padded

    const int tid   = threadIdx.x;
    const int lane  = tid & 31;
    const int wave  = tid >> 5;
    const int pix0  = (blockIdx.x * WAVES_PER_BLOCK + wave) * PIX_PER_WAVE;

    const int m     = lane & 15;           // A row (pixel within tile)
    const int halfw = lane >> 4;           // lane-half -> K sub-ranges
    const int nsel  = lane & 15;           // B column / output channel
    const int fidx  = (nsel < 3) ? nsel : 3;

    const int   prow = pix0 + m;           // tile-0 pixel for this lane
    const float px0  = (float)(prow & (IMG_W - 1));
    const float px1  = px0 + 16.0f;        // tile 1: +16 in x, same row
    const float py   = (float)(prow >> 8);

    v8f acc0 = {}, acc1 = {};

    for (int cbase = 0; cbase < NGAUSS; cbase += CHUNK) {
        __syncthreads();
        // ---- project CHUNK gaussians into LDS (4 per thread) ----
        for (int i = tid; i < CHUNK; i += TPB) {
            const int g = cbase + i;
            const float x  = tanhf(xyz[2 * g + 0]);
            const float y  = tanhf(xyz[2 * g + 1]);
            const float sx = fabsf(scaling[2 * g + 0] + 0.5f);
            const float sy = fabsf(scaling[2 * g + 1] + 0.5f);
            const float th = (1.0f / (1.0f + __expf(-rotation[g]))) * 6.28318530717958647692f;
            const float c  = __cosf(th);
            const float s  = __sinf(th);
            const float sx2 = sx * sx, sy2 = sy * sy;
            const float cov_a = c * c * sx2 + s * s * sy2;
            const float cov_b = c * s * (sx2 - sy2);
            const float cov_d = s * s * sx2 + c * c * sy2;
            const float inv   = 1.0f / (cov_a * cov_d - cov_b * cov_b);
            const float K1 = 0.72134752044448170367f;   // 0.5 / ln 2
            const float K2 = 1.44269504088896340736f;   // 1.0 / ln 2
            s_g[0][i] = 0.5f * ((x + 1.0f) * (float)IMG_W - 1.0f);
            s_g[1][i] = 0.5f * ((y + 1.0f) * (float)IMG_H - 1.0f);
            s_g[2][i] =  cov_d * inv * K1;
            s_g[3][i] = -cov_b * inv * K2;
            s_g[4][i] =  cov_a * inv * K1;
            s_g[5][i] = opacity[g];
            s_ft[0][i] = (_Float16)features[3 * g + 0];
            s_ft[1][i] = (_Float16)features[3 * g + 1];
            s_ft[2][i] = (_Float16)features[3 * g + 2];
            s_ft[3][i] = (_Float16)0.0f;
        }
        __syncthreads();

        // ---- K sweep: 32 gaussians per WMMA pair ----
        for (int kb = 0; kb < CHUNK; kb += 32) {
            // B fragment: lane-half h holds K = 16h+e -> one contiguous 32B load
            const v16h b = *(const v16h*)&s_ft[fidx][kb + 16 * halfw];
            v16h a0, a1;
            #pragma unroll
            for (int run = 0; run < 2; ++run) {
                // A K-runs: K=16h+e (contiguous, matches B)
                const int base = kb + 16 * halfw + 8 * run;
                const v8f vcx = *(const v8f*)&s_g[0][base];
                const v8f vcy = *(const v8f*)&s_g[1][base];
                const v8f vca = *(const v8f*)&s_g[2][base];
                const v8f vcb = *(const v8f*)&s_g[3][base];
                const v8f vcc = *(const v8f*)&s_g[4][base];
                const v8f vop = *(const v8f*)&s_g[5][base];
                #pragma unroll
                for (int j = 0; j < 8; ++j) {
                    const float dy  = vcy[j] - py;
                    const float t2  = vcc[j] * (dy * dy);   // shared by both tiles

                    const float dx0 = vcx[j] - px0;
                    const float s0  = fmaf(vca[j], dx0 * dx0, fmaf(vcb[j], dx0 * dy, t2));
                    float al0 = vop[j] * EXP2F(-s0);
                    al0 = (al0 >= (1.0f / 255.0f)) ? al0 : 0.0f;
                    a0[8 * run + j] = (_Float16)al0;

                    const float dx1 = vcx[j] - px1;
                    const float s1  = fmaf(vca[j], dx1 * dx1, fmaf(vcb[j], dx1 * dy, t2));
                    float al1 = vop[j] * EXP2F(-s1);
                    al1 = (al1 >= (1.0f / 255.0f)) ? al1 : 0.0f;
                    a1[8 * run + j] = (_Float16)al1;
                }
            }
            acc0 = __builtin_amdgcn_wmma_f32_16x16x32_f16(false, a0, false, b,
                                                          (short)0, acc0, false, false);
            acc1 = __builtin_amdgcn_wmma_f32_16x16x32_f16(false, a1, false, b,
                                                          (short)0, acc1, false, false);
        }
    }

    // ---- write D tiles: lane holds channel nsel, rows M = r + 8*half ----
    if (nsel < 3) {
        #pragma unroll
        for (int r = 0; r < 8; ++r) {
            const int mm = r + 8 * halfw;
            const float v0 = fminf(fmaxf(acc0[r], 0.0f), 1.0f);
            const float v1 = fminf(fmaxf(acc1[r], 0.0f), 1.0f);
            out[nsel * (IMG_H * IMG_W) + pix0 + mm]      = v0;
            out[nsel * (IMG_H * IMG_W) + pix0 + 16 + mm] = v1;
        }
    }
}

extern "C" void kernel_launch(void* const* d_in, const int* in_sizes, int n_in,
                              void* d_out, int out_size, void* d_ws, size_t ws_size,
                              hipStream_t stream) {
    (void)in_sizes; (void)n_in; (void)out_size; (void)d_ws; (void)ws_size;
    const float* xyz      = (const float*)d_in[0];
    const float* scaling  = (const float*)d_in[1];
    const float* rotation = (const float*)d_in[2];
    const float* features = (const float*)d_in[3];
    const float* opacity  = (const float*)d_in[4];
    float* out = (float*)d_out;

    const int total_waves = (IMG_H * IMG_W) / PIX_PER_WAVE;   // 2048
    const int blocks      = total_waves / WAVES_PER_BLOCK;    // 256
    gsplat_wmma_kernel<<<dim3(blocks), dim3(TPB), 0, stream>>>(
        xyz, scaling, rotation, features, opacity, out);
}
